// SAMCell_55808805044380
// MI455X (gfx1250) — matrix operand order repaired and checked by reading, not compiled, not run on hardware
//
#include <hip/hip_runtime.h>
#include <hip/hip_bf16.h>

typedef __attribute__((ext_vector_type(16))) __bf16       v16bf;
typedef __attribute__((ext_vector_type(8)))  float        v8f;
typedef __attribute__((ext_vector_type(4)))  float        v4f;
typedef __attribute__((ext_vector_type(4)))  unsigned int v4u;
typedef __attribute__((ext_vector_type(8)))  unsigned int v8u;

#define D_DIM 256
#define E_DIM 256
#define T_DIM 128
#define QIN   640          // D + E + T
#define P_DIM 8
#define KT_TILES 20        // 640 / 32
#define NT_TILES 16        // 256 / 16
#define ROWS_PER_BLOCK 64
#define THREADS 128
#define LDA 648            // padded bf16 row stride for staged A (640 + 8)
#define LDO 260            // padded f32 row stride for GEMM output staging

__device__ __forceinline__ unsigned short f32_to_bf16(float f) {
  unsigned int u = __builtin_bit_cast(unsigned int, f);
  u += 0x7fffu + ((u >> 16) & 1u);           // round-to-nearest-even
  return (unsigned short)(u >> 16);
}

__device__ __forceinline__ float waveSum(float v) {
  v += __shfl_xor(v, 16, 32);
  v += __shfl_xor(v, 8, 32);
  v += __shfl_xor(v, 4, 32);
  v += __shfl_xor(v, 2, 32);
  v += __shfl_xor(v, 1, 32);
  return v;
}

__device__ __forceinline__ float clipf(float x, float lo, float hi) {
  return fminf(fmaxf(x, lo), hi);
}

// ---------------------------------------------------------------------------
// Kernel 0: repack Wq [256 x 640] f32 into bf16 WMMA B-fragment tiles.
// Tile (kt, nt) covers K = kt*32..+31, N = nt*16..+15.
// Per-lane layout (dense 32x16 bf16 B): b[i] = B[(i<8 ? 8*g + i : 16 + 8*g + i-8)][lane&15], g=lane>>4,
// stored contiguously: idx = ((kt*16 + nt)*32 + lane)*16 + i.
// ---------------------------------------------------------------------------
__global__ void pack_wq_bf16(const float* __restrict__ Wq,
                             unsigned short* __restrict__ wsW) {
  int idx = blockIdx.x * 256 + threadIdx.x;
  if (idx >= KT_TILES * NT_TILES * 32 * 16) return;
  int i    = idx & 15;
  int lane = (idx >> 4) & 31;
  int tile = idx >> 9;
  int nt   = tile & 15;
  int kt   = tile >> 4;
  int n    = nt * 16 + (lane & 15);
  int g    = lane >> 4;
  int k    = kt * 32 + (i < 8 ? 8 * g + i : 16 + 8 * g + (i - 8));
  wsW[idx] = f32_to_bf16(Wq[n * QIN + k]);
}

// ---------------------------------------------------------------------------
// Fused SAMCell kernel: 64 rows per block, 4 waves.
// ---------------------------------------------------------------------------
__global__ __launch_bounds__(THREADS, 1)
void samcell_fused(const float* __restrict__ raw,
                   const float* __restrict__ edge,
                   const float* __restrict__ timeenc,
                   const float* __restrict__ proto,
                   const unsigned short* __restrict__ wsW,
                   const float* __restrict__ bq,
                   const float* __restrict__ Wg,
                   const float* __restrict__ bg,
                   const float* __restrict__ temperature,
                   const float* __restrict__ ln_w,
                   const float* __restrict__ ln_b,
                   float* __restrict__ out) {
  extern __shared__ char smem[];
  unsigned short* sA = (unsigned short*)smem;                        // 64 x 648 bf16
  float* sOut = (float*)(smem + ROWS_PER_BLOCK * LDA * 2);           // 64 x 260 f32

  const int t  = threadIdx.x;
  const int b0 = blockIdx.x * ROWS_PER_BLOCK;

  // ---- Phase 0: stage q_in = [raw | edge | time] as bf16 into LDS ----
  #pragma unroll 4
  for (int it = 0; it < (ROWS_PER_BLOCK * D_DIM) / (THREADS * 4); ++it) {
    int e4 = (it * THREADS + t) * 4;
    int r = e4 >> 8, c = e4 & 255;
    v4f v = *(const v4f*)(raw + (size_t)(b0 + r) * D_DIM + c);
    unsigned long long pk =  (unsigned long long)f32_to_bf16(v[0])
                          | ((unsigned long long)f32_to_bf16(v[1]) << 16)
                          | ((unsigned long long)f32_to_bf16(v[2]) << 32)
                          | ((unsigned long long)f32_to_bf16(v[3]) << 48);
    *(unsigned long long*)(sA + r * LDA + c) = pk;
  }
  #pragma unroll 4
  for (int it = 0; it < (ROWS_PER_BLOCK * E_DIM) / (THREADS * 4); ++it) {
    int e4 = (it * THREADS + t) * 4;
    int r = e4 >> 8, c = e4 & 255;
    v4f v = *(const v4f*)(edge + (size_t)(b0 + r) * E_DIM + c);
    unsigned long long pk =  (unsigned long long)f32_to_bf16(v[0])
                          | ((unsigned long long)f32_to_bf16(v[1]) << 16)
                          | ((unsigned long long)f32_to_bf16(v[2]) << 32)
                          | ((unsigned long long)f32_to_bf16(v[3]) << 48);
    *(unsigned long long*)(sA + r * LDA + D_DIM + c) = pk;
  }
  #pragma unroll 4
  for (int it = 0; it < (ROWS_PER_BLOCK * T_DIM) / (THREADS * 4); ++it) {
    int e4 = (it * THREADS + t) * 4;
    int r = e4 >> 7, c = e4 & 127;
    v4f v = *(const v4f*)(timeenc + (size_t)(b0 + r) * T_DIM + c);
    unsigned long long pk =  (unsigned long long)f32_to_bf16(v[0])
                          | ((unsigned long long)f32_to_bf16(v[1]) << 16)
                          | ((unsigned long long)f32_to_bf16(v[2]) << 32)
                          | ((unsigned long long)f32_to_bf16(v[3]) << 48);
    *(unsigned long long*)(sA + r * LDA + (D_DIM + E_DIM) + c) = pk;
  }
  __syncthreads();

  // ---- Phase 1: GEMM  C[64x256] = A[64x640](bf16) x Wq^T(bf16), f32 acc ----
  const int lane = t & 31;
  const int w    = t >> 5;          // wave id: rows 16w..16w+15
  const int g    = lane >> 4;
  const int nlo  = lane & 15;

  // A fragment base for this lane: row = 16w + (lane&15), K offset 8g.
  const unsigned short* aBase = sA + (16 * w + nlo) * LDA + 8 * g;

  for (int nt = 0; nt < NT_TILES; ++nt) {
    v8f acc = {0.f, 0.f, 0.f, 0.f, 0.f, 0.f, 0.f, 0.f};
    #pragma unroll 4
    for (int kt = 0; kt < KT_TILES; ++kt) {
      // A 16x32 bf16 fragment: elements 0..7 -> K = kt*32+8g+i ;
      //                        elements 8..15 -> K = kt*32+16+8g+(i-8)
      v4u a0 = *(const v4u*)(aBase + kt * 32);
      v4u a1 = *(const v4u*)(aBase + kt * 32 + 16);
      const unsigned short* bp = wsW + (((size_t)kt * 16 + nt) * 32 + lane) * 16;
      v4u b_0 = *(const v4u*)(bp);
      v4u b_1 = *(const v4u*)(bp + 8);
      v8u au = __builtin_shufflevector(a0, a1, 0, 1, 2, 3, 4, 5, 6, 7);
      v8u bu = __builtin_shufflevector(b_0, b_1, 0, 1, 2, 3, 4, 5, 6, 7);
      v16bf av = __builtin_bit_cast(v16bf, au);
      v16bf bv = __builtin_bit_cast(v16bf, bu);
      acc = __builtin_amdgcn_wmma_f32_16x16x32_bf16(
          false, av, false, bv, (short)0, acc, false, false);
    }
    float bias = bq[nt * 16 + nlo];
    #pragma unroll
    for (int r = 0; r < 8; ++r) {   // C layout: M = r + 8g, N = lane&15
      sOut[(16 * w + r + 8 * g) * LDO + nt * 16 + nlo] = acc[r] + bias;
    }
  }
  __syncthreads();

  // ---- Phase 2: per-row epilogue, lane-parallel over d = 4*lane + 128*jj ----
  v4f lnw[2], lnb[2], wgr[2], wgc[2];
  #pragma unroll
  for (int jj = 0; jj < 2; ++jj) {
    int d = 4 * lane + 128 * jj;
    lnw[jj] = *(const v4f*)(ln_w + d);
    lnb[jj] = *(const v4f*)(ln_b + d);
    wgr[jj] = *(const v4f*)(Wg + d);
    wgc[jj] = *(const v4f*)(Wg + D_DIM + d);
  }
  const v4f wgt = *(const v4f*)(Wg + 2 * D_DIM + 4 * lane);
  const float invTemp = 1.0f / (clipf(temperature[0], 0.5f, 5.0f) + 0.01f);
  const float bgv = bg[0];

  for (int rr = 0; rr < 16; ++rr) {
    const int row = 16 * w + rr;
    const size_t b = (size_t)(b0 + row);

    // LayerNorm1 + tanh -> query
    v4f x[2];
    float s = 0.f;
    #pragma unroll
    for (int jj = 0; jj < 2; ++jj) {
      x[jj] = *(const v4f*)(sOut + row * LDO + 4 * lane + 128 * jj);
      #pragma unroll
      for (int i = 0; i < 4; ++i) s += x[jj][i];
    }
    const float mu = waveSum(s) * (1.f / 256.f);
    float vs = 0.f;
    #pragma unroll
    for (int jj = 0; jj < 2; ++jj)
      #pragma unroll
      for (int i = 0; i < 4; ++i) { float d = x[jj][i] - mu; vs += d * d; }
    const float rstd = rsqrtf(waveSum(vs) * (1.f / 256.f) + 1e-6f);
    v4f q[2];
    float qs = 0.f;
    #pragma unroll
    for (int jj = 0; jj < 2; ++jj)
      #pragma unroll
      for (int i = 0; i < 4; ++i) {
        float qq = tanhf((x[jj][i] - mu) * rstd * lnw[jj][i] + lnb[jj][i]);
        q[jj][i] = qq;
        qs += qq * qq;
      }
    const float qinv = 1.0f / fmaxf(sqrtf(waveSum(qs)), 1e-6f);

    // cosine similarity to 8 prototypes (kept in registers for reuse)
    const float* pb = proto + b * (size_t)(P_DIM * D_DIM);
    v4f pr[P_DIM][2];
    float attn[P_DIM], smax = -1e30f;
    #pragma unroll
    for (int p = 0; p < P_DIM; ++p) {
      float dt = 0.f, ss = 0.f;
      #pragma unroll
      for (int jj = 0; jj < 2; ++jj) {
        v4f pv = *(const v4f*)(pb + p * D_DIM + 4 * lane + 128 * jj);
        pr[p][jj] = pv;
        #pragma unroll
        for (int i = 0; i < 4; ++i) {
          dt += q[jj][i] * pv[i];
          ss += pv[i] * pv[i];
        }
      }
      dt = waveSum(dt);
      ss = waveSum(ss);
      float pinv = 1.0f / fmaxf(sqrtf(ss), 1e-6f);
      float sim = clipf(dt * qinv * pinv, -10.f, 10.f) * invTemp;
      attn[p] = sim;
      smax = fmaxf(smax, sim);
    }
    float esum = 0.f;
    #pragma unroll
    for (int p = 0; p < P_DIM; ++p) { attn[p] = expf(attn[p] - smax); esum += attn[p]; }
    const float einv = 1.0f / esum;
    #pragma unroll
    for (int p = 0; p < P_DIM; ++p) attn[p] = clipf(attn[p] * einv, 0.f, 1.f);

    v4f cand[2];
    #pragma unroll
    for (int jj = 0; jj < 2; ++jj)
      #pragma unroll
      for (int i = 0; i < 4; ++i) {
        float c = 0.f;
        #pragma unroll
        for (int p = 0; p < P_DIM; ++p) c += attn[p] * pr[p][jj][i];
        cand[jj][i] = clipf(c, -5.f, 5.f);
      }

    // gate = sigmoid(clip(g_in . Wg + bg))
    v4f rv[2];
    float gdot = 0.f;
    #pragma unroll
    for (int jj = 0; jj < 2; ++jj) {
      rv[jj] = *(const v4f*)(raw + b * D_DIM + 4 * lane + 128 * jj);
      #pragma unroll
      for (int i = 0; i < 4; ++i) {
        gdot += clipf(rv[jj][i], -50.f, 50.f) * wgr[jj][i]
              + clipf(cand[jj][i], -50.f, 50.f) * wgc[jj][i];
      }
    }
    {
      v4f tv = *(const v4f*)(timeenc + b * T_DIM + 4 * lane);
      #pragma unroll
      for (int i = 0; i < 4; ++i) gdot += clipf(tv[i], -50.f, 50.f) * wgt[i];
    }
    float gpre = clipf(waveSum(gdot) + bgv, -10.f, 10.f);
    const float gate = 1.0f / (1.0f + expf(-gpre));

    // gated blend + LayerNorm2 + clip
    v4f u[2];
    float us = 0.f;
    #pragma unroll
    for (int jj = 0; jj < 2; ++jj)
      #pragma unroll
      for (int i = 0; i < 4; ++i) {
        float uu = (1.f - gate) * rv[jj][i] + gate * cand[jj][i];
        u[jj][i] = uu;
        us += uu;
      }
    const float mu2 = waveSum(us) * (1.f / 256.f);
    float vs2 = 0.f;
    #pragma unroll
    for (int jj = 0; jj < 2; ++jj)
      #pragma unroll
      for (int i = 0; i < 4; ++i) { float d = u[jj][i] - mu2; vs2 += d * d; }
    const float rstd2 = rsqrtf(waveSum(vs2) * (1.f / 256.f) + 1e-6f);
    #pragma unroll
    for (int jj = 0; jj < 2; ++jj) {
      v4f o;
      #pragma unroll
      for (int i = 0; i < 4; ++i)
        o[i] = clipf((u[jj][i] - mu2) * rstd2 * lnw[jj][i] + lnb[jj][i], -10.f, 10.f);
      *(v4f*)(out + b * D_DIM + 4 * lane + 128 * jj) = o;
    }
  }
}

extern "C" void kernel_launch(void* const* d_in, const int* in_sizes, int n_in,
                              void* d_out, int out_size, void* d_ws, size_t ws_size,
                              hipStream_t stream) {
  const float* raw   = (const float*)d_in[0];
  // d_in[1] = node_features (unused by reference)
  const float* edge  = (const float*)d_in[2];
  const float* timee = (const float*)d_in[3];
  const float* proto = (const float*)d_in[4];
  const float* Wq    = (const float*)d_in[5];
  const float* bq    = (const float*)d_in[6];
  const float* Wg    = (const float*)d_in[7];
  const float* bg    = (const float*)d_in[8];
  const float* temp  = (const float*)d_in[9];
  const float* lnw   = (const float*)d_in[10];
  const float* lnb   = (const float*)d_in[11];
  float* out = (float*)d_out;
  unsigned short* wsW = (unsigned short*)d_ws;  // 320 KB bf16 Wq tiles

  const int B = in_sizes[0] / D_DIM;

  const int packElems = KT_TILES * NT_TILES * 32 * 16;
  pack_wq_bf16<<<(packElems + 255) / 256, 256, 0, stream>>>(Wq, wsW);

  const size_t shbytes = (size_t)ROWS_PER_BLOCK * LDA * 2 +
                         (size_t)ROWS_PER_BLOCK * LDO * 4;  // 149,504 B
  samcell_fused<<<B / ROWS_PER_BLOCK, THREADS, shbytes, stream>>>(
      raw, edge, timee, proto, wsW, bq, Wg, bg, temp, lnw, lnb, out);
}
